// CFGCN_7310034338177
// MI455X (gfx1250) — matrix operand order repaired and checked by reading, not compiled, not run on hardware
//
#include <hip/hip_runtime.h>
#include <math.h>

#define N_NODES 200000
#define D 64
#define D4 (D/4)
#define N_EDGES 1250000
#define BATCH 16384
#define LAM 0.001f

typedef __attribute__((ext_vector_type(16))) _Float16 v16h;
typedef __attribute__((ext_vector_type(8)))  float    v8f;

// ---------------- helpers ----------------

__device__ __forceinline__ float pick8(v8f c, int idx) {
    float r = c[0];
#pragma unroll
    for (int j = 1; j < 8; ++j) r = (idx == j) ? c[j] : r;
    return r;
}

// ---------------- kernels ----------------

__global__ void zero_accum_kernel(float* accums) {
    if (threadIdx.x < 2) accums[threadIdx.x] = 0.0f;
}

// acc = emb ; h_scaled = emb * sd ; h_next = 0
__global__ void init_kernel(const float* __restrict__ emb,
                            const float* __restrict__ sd,
                            float* __restrict__ acc,
                            float* __restrict__ hsc,
                            float* __restrict__ hnx) {
    int i = blockIdx.x * blockDim.x + threadIdx.x;       // float4 index
    if (i >= N_NODES * D4) return;
    int node = i >> 4;                                   // D4 == 16
    float s = sd[node];
    float4 v = ((const float4*)emb)[i];
    ((float4*)acc)[i] = v;
    float4 vs; vs.x = v.x * s; vs.y = v.y * s; vs.z = v.z * s; vs.w = v.w * s;
    ((float4*)hsc)[i] = vs;
    float4 z; z.x = 0.f; z.y = 0.f; z.z = 0.f; z.w = 0.f;
    ((float4*)hnx)[i] = z;
}

// h_next[dst] += h_scaled[src]   (16 threads per edge, float4 each)
__global__ void scatter_kernel(const int* __restrict__ src,
                               const int* __restrict__ dst,
                               const float* __restrict__ hsc,
                               float* __restrict__ hnx) {
    long long t = (long long)blockIdx.x * blockDim.x + threadIdx.x;
    if (t >= (long long)N_EDGES * 16) return;
    int e = (int)(t >> 4);
    int c = (int)(t & 15);
    int s = src[e];
    int d = dst[e];
    float4 v = ((const float4*)hsc)[(size_t)s * D4 + c];
    float* out = hnx + (size_t)d * D + c * 4;
    atomicAdd(out + 0, v.x);
    atomicAdd(out + 1, v.y);
    atomicAdd(out + 2, v.z);
    atomicAdd(out + 3, v.w);
}

// h = h_next * sd ; acc += h ; h_scaled = h * sd ; h_next = 0 (for next layer)
__global__ void combine_kernel(const float* __restrict__ sd,
                               float* __restrict__ acc,
                               float* __restrict__ hsc,
                               float* __restrict__ hnx) {
    int i = blockIdx.x * blockDim.x + threadIdx.x;
    if (i >= N_NODES * D4) return;
    int node = i >> 4;
    float s = sd[node];
    float4 hv = ((float4*)hnx)[i];
    float4 z; z.x = 0.f; z.y = 0.f; z.z = 0.f; z.w = 0.f;
    ((float4*)hnx)[i] = z;
    hv.x *= s; hv.y *= s; hv.z *= s; hv.w *= s;
    float4 a = ((float4*)acc)[i];
    a.x += hv.x; a.y += hv.y; a.z += hv.z; a.w += hv.w;
    ((float4*)acc)[i] = a;
    hv.x *= s; hv.y *= s; hv.z *= s; hv.w *= s;
    ((float4*)hsc)[i] = hv;
}

// reg_loss = sum(emb[u]^2 + emb[p]^2 + emb[n]^2) over the batch
__global__ void reg_kernel(const float* __restrict__ emb,
                           const int* __restrict__ users,
                           const int* __restrict__ pos,
                           const int* __restrict__ neg,
                           const int* __restrict__ nu_p,
                           float* __restrict__ accums) {
    int b = blockIdx.x * blockDim.x + threadIdx.x;
    if (b >= BATCH) return;
    int nu = nu_p[0];
    const float4* u = (const float4*)(emb + (size_t)users[b] * D);
    const float4* p = (const float4*)(emb + (size_t)(pos[b] + nu) * D);
    const float4* n = (const float4*)(emb + (size_t)(neg[b] + nu) * D);
    float sum = 0.f;
#pragma unroll
    for (int j = 0; j < D4; ++j) {
        float4 a = u[j]; sum += a.x*a.x + a.y*a.y + a.z*a.z + a.w*a.w;
        float4 q = p[j]; sum += q.x*q.x + q.y*q.y + q.z*q.z + q.w*q.w;
        float4 r = n[j]; sum += r.x*r.x + r.y*r.y + r.z*r.z + r.w*r.w;
    }
    atomicAdd(&accums[1], sum);
}

// WMMA scoring: one wave32 per 16 batch rows.
// pos/neg scores = diag( (U/4) * (P/4)^T ) via v_wmma_f32_16x16x32_f16, K=64 in 2 chunks.
__global__ void score_kernel(const float* __restrict__ acc,
                             const int* __restrict__ users,
                             const int* __restrict__ pos,
                             const int* __restrict__ neg,
                             const int* __restrict__ nu_p,
                             float* __restrict__ accums) {
    int wave = (blockIdx.x * blockDim.x + threadIdx.x) >> 5; // tile id, exactly BATCH/16 waves
    int lane = threadIdx.x & 31;
    int nu = nu_p[0];

    int m  = lane & 15;   // A-row / B-col index held by this lane
    int hl = lane >> 4;   // which half of the wave

    const float* urow = acc + (size_t)users[wave * 16 + m] * D;
    const float* prow = acc + (size_t)(pos[wave * 16 + m] + nu) * D;
    const float* nrow = acc + (size_t)(neg[wave * 16 + m] + nu) * D;

    const float scale = 0.25f; // prop = acc/4, folded into operands

    v16h a0, a1, bp0, bp1, bn0, bn1;
#pragma unroll
    for (int j = 0; j < 8; ++j) {
        // A (16x32, 16-bit): lane half hl, VGPR j -> K = g*16 + hl*8 + w*2 (g=j>>2, w=j&3)
        int g = j >> 2, w = j & 3;
        int ka = g * 16 + hl * 8 + w * 2;
        // B (32x16, 16-bit): same K mapping as A
        int kb = ka;

        a0[2*j]   = (_Float16)(urow[ka]        * scale);
        a0[2*j+1] = (_Float16)(urow[ka + 1]    * scale);
        a1[2*j]   = (_Float16)(urow[32 + ka]   * scale);
        a1[2*j+1] = (_Float16)(urow[32 + ka+1] * scale);

        bp0[2*j]   = (_Float16)(prow[kb]        * scale);
        bp0[2*j+1] = (_Float16)(prow[kb + 1]    * scale);
        bp1[2*j]   = (_Float16)(prow[32 + kb]   * scale);
        bp1[2*j+1] = (_Float16)(prow[32 + kb+1] * scale);

        bn0[2*j]   = (_Float16)(nrow[kb]        * scale);
        bn0[2*j+1] = (_Float16)(nrow[kb + 1]    * scale);
        bn1[2*j]   = (_Float16)(nrow[32 + kb]   * scale);
        bn1[2*j+1] = (_Float16)(nrow[32 + kb+1] * scale);
    }

    v8f cp = {}; v8f cn = {};
    cp = __builtin_amdgcn_wmma_f32_16x16x32_f16(false, a0, false, bp0, (short)0, cp, false, false);
    cp = __builtin_amdgcn_wmma_f32_16x16x32_f16(false, a1, false, bp1, (short)0, cp, false, false);
    cn = __builtin_amdgcn_wmma_f32_16x16x32_f16(false, a0, false, bn0, (short)0, cn, false, false);
    cn = __builtin_amdgcn_wmma_f32_16x16x32_f16(false, a1, false, bn1, (short)0, cn, false, false);

    // Diagonal D[m][m]: m<8 -> lane m, VGPR m ; m>=8 -> lane m+16, VGPR m-8
    bool valid = (lane < 8) || (lane >= 24);
    int idx = (lane < 8) ? lane : (lane - 24);
    float ps = pick8(cp, idx);
    float ns = pick8(cn, idx);
    if (valid) {
        float x = ns - ps;
        float sp = fmaxf(x, 0.f) + log1pf(__expf(-fabsf(x))); // stable softplus
        atomicAdd(&accums[0], sp);
    }
}

__global__ void finalize_kernel(const float* __restrict__ accums, float* __restrict__ out) {
    if (blockIdx.x == 0 && threadIdx.x == 0) {
        float inv = 1.0f / (float)BATCH;
        out[0] = accums[0] * inv + LAM * 0.5f * accums[1] * inv;
    }
}

// ---------------- launch ----------------

extern "C" void kernel_launch(void* const* d_in, const int* in_sizes, int n_in,
                              void* d_out, int out_size, void* d_ws, size_t ws_size,
                              hipStream_t stream) {
    const float* emb = (const float*)d_in[0];
    const float* sd  = (const float*)d_in[1];
    const int* src   = (const int*)d_in[2];
    const int* dst   = (const int*)d_in[3];
    const int* users = (const int*)d_in[4];
    const int* pos   = (const int*)d_in[5];
    const int* neg   = (const int*)d_in[6];
    const int* nu    = (const int*)d_in[7];

    const size_t nodeBytes = (size_t)N_NODES * D * sizeof(float); // 51.2 MB, 256B aligned
    char* ws = (char*)d_ws;
    float* acc    = (float*)(ws);
    float* hsc    = (float*)(ws + nodeBytes);
    float* hnx    = (float*)(ws + 2 * nodeBytes);
    float* accums = (float*)(ws + 3 * nodeBytes); // [0]=loss sum, [1]=reg sum

    zero_accum_kernel<<<1, 32, 0, stream>>>(accums);

    const int n4 = N_NODES * D4; // 3,200,000 float4 elements
    init_kernel<<<(n4 + 255) / 256, 256, 0, stream>>>(emb, sd, acc, hsc, hnx);

    const long long st = (long long)N_EDGES * 16;
    const int sblocks = (int)((st + 255) / 256);
    for (int l = 0; l < 3; ++l) {
        scatter_kernel<<<sblocks, 256, 0, stream>>>(src, dst, hsc, hnx);
        combine_kernel<<<(n4 + 255) / 256, 256, 0, stream>>>(sd, acc, hsc, hnx);
    }

    reg_kernel<<<(BATCH + 255) / 256, 256, 0, stream>>>(emb, users, pos, neg, nu, accums);

    // BATCH/16 = 1024 waves, 8 waves (256 threads) per block -> 128 blocks, no remainder
    score_kernel<<<(BATCH / 16) / 8, 256, 0, stream>>>(acc, users, pos, neg, nu, accums);

    finalize_kernel<<<1, 1, 0, stream>>>(accums, (float*)d_out);
}
